// ParallelBlock_11854109737663
// MI455X (gfx1250) — matrix operand order repaired and checked by reading, not compiled, not run on hardware
//
#include <hip/hip_runtime.h>

// ---------------------------------------------------------------------------
// Types for CDNA5 WMMA (wave32)
// ---------------------------------------------------------------------------
typedef __attribute__((ext_vector_type(16))) __bf16        v16bf;
typedef __attribute__((ext_vector_type(8)))  float         v8f;
typedef __attribute__((ext_vector_type(4)))  unsigned int  u32x4;
typedef __attribute__((ext_vector_type(8)))  unsigned int  u32x8;

static __device__ __forceinline__ v16bf frag_from(u32x4 lo, u32x4 hi) {
    u32x8 t;
    t.lo = lo;
    t.hi = hi;
    return __builtin_bit_cast(v16bf, t);
}

// bf16 <-> f32 (round-to-nearest-even)
static __device__ __forceinline__ unsigned short f2bf(float f) {
    unsigned int u = __float_as_uint(f);
    unsigned int r = u + 0x7FFFu + ((u >> 16) & 1u);
    return (unsigned short)(r >> 16);
}
static __device__ __forceinline__ float bf2f(unsigned short h) {
    return __uint_as_float(((unsigned int)h) << 16);
}

// A-fragment (16x32, M x K), row-major source [row][k], ld in elements.
// Lane l: row = m0 + (l&15); K set = {ko..ko+7, 16+ko..16+ko+7}, ko = (l>=16)?8:0
static __device__ __forceinline__ v16bf load_a(const unsigned short* __restrict__ p,
                                               int ld, int m0, int kb, int lane) {
    int row = m0 + (lane & 15);
    int ko  = (lane >= 16) ? 8 : 0;
    const unsigned short* q = p + row * ld + kb + ko;
    u32x4 lo = *(const u32x4*)(q);
    u32x4 hi = *(const u32x4*)(q + 16);
    return frag_from(lo, hi);
}

// B-fragment (32x16, K x N), source row-major in K per output column: [col][k].
// Lane l: col = n0 + (l&15); K set = {ko..ko+7, 16+ko..16+ko+7}, ko = (l>=16)?8:0
static __device__ __forceinline__ v16bf load_b(const unsigned short* __restrict__ p,
                                               int ld, int n0, int kb, int lane) {
    int col = n0 + (lane & 15);
    int ko  = (lane >= 16) ? 8 : 0;
    const unsigned short* q = p + col * ld + kb + ko;
    u32x4 lo = *(const u32x4*)(q);
    u32x4 hi = *(const u32x4*)(q + 16);
    return frag_from(lo, hi);
}

static __device__ __forceinline__ v8f wmma_bf16(v16bf a, v16bf b, v8f c) {
    return __builtin_amdgcn_wmma_f32_16x16x32_bf16(false, a, false, b, (short)0, c,
                                                   false, false);
}

// ---------------------------------------------------------------------------
// Problem constants
// ---------------------------------------------------------------------------
#define BN     8
#define NN     1024
#define CC     1024
#define HH     16
#define DD     64
#define HID    4096
#define MM     (BN * NN)        // 8192
#define OIN    (HID + 3 * CC)   // 7168
#define KOUT   (HID + CC)       // 5120

// ---------------------------------------------------------------------------
// K0a: f32 -> bf16 weight conversion (in_proj_w, 7168x1024)
// ---------------------------------------------------------------------------
__global__ void __launch_bounds__(256)
cvt_bf16_kernel(const float* __restrict__ src, unsigned short* __restrict__ dst, int n) {
    for (int i = blockIdx.x * 256 + threadIdx.x; i < n; i += gridDim.x * 256)
        dst[i] = f2bf(src[i]);
}

// K0b: Wsum[o][k] = bf16(Wout[o][k] + Wout[o+C][k]),  o<1024, k<5120
__global__ void __launch_bounds__(256)
wsum_kernel(const float* __restrict__ w, unsigned short* __restrict__ dst) {
    int i = blockIdx.x * 256 + threadIdx.x;
    if (i < CC * KOUT)
        dst[i] = f2bf(w[i] + w[i + CC * KOUT]);
}

// ---------------------------------------------------------------------------
// K1: input LayerNorm over C=1024, one workgroup per row, output bf16
// ---------------------------------------------------------------------------
__global__ void __launch_bounds__(256)
ln_in_kernel(const float* __restrict__ x, const float* __restrict__ w,
             const float* __restrict__ b, unsigned short* __restrict__ y) {
    int row = blockIdx.x;
    const float* xr = x + row * CC;
    float vals[4];
    float s = 0.f, s2 = 0.f;
#pragma unroll
    for (int i = 0; i < 4; i++) {
        float f = xr[threadIdx.x + 256 * i];
        vals[i] = f;
        s += f;
        s2 += f * f;
    }
#pragma unroll
    for (int m = 1; m < 32; m <<= 1) {
        s  += __shfl_xor(s, m, 32);
        s2 += __shfl_xor(s2, m, 32);
    }
    __shared__ float ss[8], ss2[8];
    if ((threadIdx.x & 31) == 0) {
        ss[threadIdx.x >> 5]  = s;
        ss2[threadIdx.x >> 5] = s2;
    }
    __syncthreads();
    float ts = 0.f, ts2 = 0.f;
#pragma unroll
    for (int i = 0; i < 8; i++) { ts += ss[i]; ts2 += ss2[i]; }
    float mean = ts * (1.0f / CC);
    float var  = ts2 * (1.0f / CC) - mean * mean;
    float rinv = rsqrtf(var + 1e-5f);
#pragma unroll
    for (int i = 0; i < 4; i++) {
        int c = threadIdx.x + 256 * i;
        y[row * CC + c] = f2bf((vals[i] - mean) * rinv * w[c] + b[c]);
    }
}

// ---------------------------------------------------------------------------
// K2: z = LN(x) @ Win^T (+bias), fused epilogue.
//   M=8192, K=1024, O=7168.  Block = 128x256 (8 waves, each 64x64 -> 16 WMMA tiles).
//   bias[o] = (o>=3072) ? mlp_bias[o-3072] : 0   (matches reference concat order)
//   o in [0,4096):      GELU -> mlp_act bf16
//   o in [4096,5120):   qbuf bf16 (raw, pre-LN)
//   o in [5120,6144):   kbuf bf16
//   o in [6144,7168):   v scattered to vT (B,H,D,N) bf16
// ---------------------------------------------------------------------------
__global__ void __launch_bounds__(256)
gemm_in_kernel(const unsigned short* __restrict__ y,
               const unsigned short* __restrict__ w,
               const float* __restrict__ mlp_bias,
               unsigned short* __restrict__ mlp_act,
               unsigned short* __restrict__ qbuf,
               unsigned short* __restrict__ kbuf,
               unsigned short* __restrict__ vT) {
    int lane = threadIdx.x & 31;
    int wid  = threadIdx.x >> 5;
    int wm = wid >> 2, wn = wid & 3;
    int m0 = blockIdx.x * 128 + wm * 64;
    int n0 = blockIdx.y * 256 + wn * 64;

    v8f zero = {0.f, 0.f, 0.f, 0.f, 0.f, 0.f, 0.f, 0.f};
    v8f acc[4][4];
#pragma unroll
    for (int i = 0; i < 4; i++)
#pragma unroll
        for (int j = 0; j < 4; j++) acc[i][j] = zero;

    for (int kb = 0; kb < CC; kb += 32) {
        if (kb + 64 < CC) {
            __builtin_prefetch(y + (m0 + (lane & 15)) * CC + kb + 64, 0, 3);
            __builtin_prefetch(w + (n0 + (lane & 15)) * CC + kb + 64, 0, 3);
        }
        v16bf af[4], bf_[4];
#pragma unroll
        for (int i = 0; i < 4; i++) af[i]  = load_a(y, CC, m0 + 16 * i, kb, lane);
#pragma unroll
        for (int j = 0; j < 4; j++) bf_[j] = load_b(w, CC, n0 + 16 * j, kb, lane);
#pragma unroll
        for (int i = 0; i < 4; i++)
#pragma unroll
            for (int j = 0; j < 4; j++) acc[i][j] = wmma_bf16(af[i], bf_[j], acc[i][j]);
    }

    int half = (lane >= 16) ? 8 : 0;
    int ncol = lane & 15;
#pragma unroll
    for (int j = 0; j < 4; j++) {
        int o = n0 + 16 * j + ncol;
        float bias = (o >= 3 * CC) ? mlp_bias[o - 3 * CC] : 0.0f;
#pragma unroll
        for (int i = 0; i < 4; i++) {
#pragma unroll
            for (int r = 0; r < 8; r++) {
                int m = m0 + 16 * i + r + half;
                float v = acc[i][j][r] + bias;
                if (o < HID) {
                    float g = 0.5f * v * (1.0f + erff(v * 0.70710678118654752f));
                    mlp_act[m * HID + o] = f2bf(g);
                } else if (o < HID + CC) {
                    qbuf[m * CC + (o - HID)] = f2bf(v);
                } else if (o < HID + 2 * CC) {
                    kbuf[m * CC + (o - HID - CC)] = f2bf(v);
                } else {
                    int b = m >> 10, n = m & 1023, hd = o - (HID + 2 * CC);
                    vT[(b * CC + hd) * NN + n] = f2bf(v);
                }
            }
        }
    }
}

// ---------------------------------------------------------------------------
// K3: per-head LN of q,k over D=64, write (B,H,N,D) bf16. One wave per
// (row, head, q|k); 2 elements per lane.
// ---------------------------------------------------------------------------
__global__ void __launch_bounds__(256)
qk_ln_kernel(const unsigned short* __restrict__ qbuf,
             const unsigned short* __restrict__ kbuf,
             const float* __restrict__ qw, const float* __restrict__ qb,
             const float* __restrict__ kw, const float* __restrict__ kb,
             unsigned short* __restrict__ qT, unsigned short* __restrict__ kT) {
    int gw   = blockIdx.x * 8 + (threadIdx.x >> 5);
    int lane = threadIdx.x & 31;
    int which = gw & 1;
    int t = gw >> 1;
    int h = t & 15;
    int m = t >> 4;
    const unsigned short* src = which ? kbuf : qbuf;
    const float* wgt = which ? kw : qw;
    const float* bia = which ? kb : qb;
    unsigned short* dst = which ? kT : qT;

    const unsigned short* p = src + m * CC + h * DD;
    float v0 = bf2f(p[lane]);
    float v1 = bf2f(p[lane + 32]);
    float s = v0 + v1, s2 = v0 * v0 + v1 * v1;
#pragma unroll
    for (int mm = 1; mm < 32; mm <<= 1) {
        s  += __shfl_xor(s, mm, 32);
        s2 += __shfl_xor(s2, mm, 32);
    }
    float mean = s * (1.0f / DD);
    float var  = s2 * (1.0f / DD) - mean * mean;
    float r    = rsqrtf(var + 1e-5f);
    int b = m >> 10, n = m & 1023;
    unsigned short* d = dst + ((b * HH + h) * NN + n) * DD;
    d[lane]      = f2bf((v0 - mean) * r * wgt[lane]      + bia[lane]);
    d[lane + 32] = f2bf((v1 - mean) * r * wgt[lane + 32] + bia[lane + 32]);
}

// ---------------------------------------------------------------------------
// K4: flash attention. One wave = 16 query rows of one (b,h); loop over key
// chunks of 64. q@k^T and P@v via WMMA bf16; online softmax with 16-lane
// shuffle reductions; P transposed C-layout -> A-layout through private LDS.
//   qT,kT: (B*H, N, D) row-major;  vT: (B*H, D, N) row-major.
//   Output attn: (B*N, C) with column = h*64+d (i.e. (B,N,H,D)).
// ---------------------------------------------------------------------------
__global__ void __launch_bounds__(256)
attn_kernel(const unsigned short* __restrict__ qT,
            const unsigned short* __restrict__ kT,
            const unsigned short* __restrict__ vT,
            unsigned short* __restrict__ attn) {
    __shared__ __align__(16) unsigned short pbuf[8][16 * 64];
    int wid  = threadIdx.x >> 5;
    int lane = threadIdx.x & 31;
    int bh   = blockIdx.x >> 3;                       // 0..127
    int qblk = (blockIdx.x & 7) * 128 + wid * 16;     // query row base

    const unsigned short* qp = qT + (size_t)bh * NN * DD;
    const unsigned short* kp = kT + (size_t)bh * NN * DD;
    const unsigned short* vp = vT + (size_t)bh * DD * NN;

    v16bf aq0 = load_a(qp, DD, qblk, 0, lane);
    v16bf aq1 = load_a(qp, DD, qblk, 32, lane);

    v8f zero = {0.f, 0.f, 0.f, 0.f, 0.f, 0.f, 0.f, 0.f};
    v8f o[4];
#pragma unroll
    for (int j = 0; j < 4; j++) o[j] = zero;
    float mrun[8], lrun[8];
#pragma unroll
    for (int r = 0; r < 8; r++) { mrun[r] = -1e30f; lrun[r] = 0.f; }

    const float scale = 0.125f;   // 1/sqrt(64)
    int half = (lane >= 16) ? 8 : 0;
    int ncol = lane & 15;
    unsigned short* pl = &pbuf[wid][0];

    for (int kc = 0; kc < NN; kc += 64) {
        // S = q @ k^T for 64 keys (4 col tiles x 2 k-steps)
        v8f s[4];
#pragma unroll
        for (int j = 0; j < 4; j++) {
            v16bf b0 = load_b(kp, DD, kc + 16 * j, 0, lane);
            v16bf b1 = load_b(kp, DD, kc + 16 * j, 32, lane);
            v8f sz = zero;
            sz = wmma_bf16(aq0, b0, sz);
            sz = wmma_bf16(aq1, b1, sz);
            s[j] = sz;
        }
        // online softmax per row (rows live across 16-lane halves)
#pragma unroll
        for (int r = 0; r < 8; r++) {
            float mx = -1e30f;
#pragma unroll
            for (int j = 0; j < 4; j++) mx = fmaxf(mx, s[j][r]);
#pragma unroll
            for (int mm = 1; mm < 16; mm <<= 1) mx = fmaxf(mx, __shfl_xor(mx, mm, 32));
            mx *= scale;
            float mnew = fmaxf(mrun[r], mx);
            float a = __expf(mrun[r] - mnew);
            float lsum = 0.f;
#pragma unroll
            for (int j = 0; j < 4; j++) {
                float p = __expf(s[j][r] * scale - mnew);
                s[j][r] = p;
                lsum += p;
            }
#pragma unroll
            for (int mm = 1; mm < 16; mm <<= 1) lsum += __shfl_xor(lsum, mm, 32);
            lrun[r] = lrun[r] * a + lsum;
            mrun[r] = mnew;
#pragma unroll
            for (int j = 0; j < 4; j++) o[j][r] *= a;
        }
        // P (C-layout) -> LDS row-major 16x64 bf16
#pragma unroll
        for (int j = 0; j < 4; j++)
#pragma unroll
            for (int r = 0; r < 8; r++)
                pl[(r + half) * 64 + 16 * j + ncol] = f2bf(s[j][r]);
        asm volatile("s_wait_dscnt 0" ::: "memory");
        // reload as A-fragments and accumulate O += P @ V
        v16bf pa0 = load_a(pl, 64, 0, 0, lane);
        v16bf pa1 = load_a(pl, 64, 0, 32, lane);
#pragma unroll
        for (int j = 0; j < 4; j++) {
            v16bf vb0 = load_b(vp, NN, 16 * j, kc, lane);
            v16bf vb1 = load_b(vp, NN, 16 * j, kc + 32, lane);
            o[j] = wmma_bf16(pa0, vb0, o[j]);
            o[j] = wmma_bf16(pa1, vb1, o[j]);
        }
        asm volatile("s_wait_dscnt 0" ::: "memory");
    }

    int b = bh >> 4, h = bh & 15;
#pragma unroll
    for (int j = 0; j < 4; j++) {
#pragma unroll
        for (int r = 0; r < 8; r++) {
            int qrow = qblk + r + half;
            float val = o[j][r] / lrun[r];
            int d = 16 * j + ncol;
            attn[((size_t)(b * NN + qrow)) * CC + h * DD + d] = f2bf(val);
        }
    }
}

// ---------------------------------------------------------------------------
// K5: out = [gelu(mlp) | attn] @ Wsum^T;  final = x + gamma*(out + bsum)
//   M=8192, K=5120 (4096 from mlp_act, 1024 from attn), O=1024.
// ---------------------------------------------------------------------------
__global__ void __launch_bounds__(256)
gemm_out_kernel(const unsigned short* __restrict__ mlp_act,
                const unsigned short* __restrict__ attn,
                const unsigned short* __restrict__ wsum,
                const float* __restrict__ out_b,
                const float* __restrict__ ls_gamma,
                const float* __restrict__ x,
                float* __restrict__ out) {
    int lane = threadIdx.x & 31;
    int wid  = threadIdx.x >> 5;
    int wm = wid >> 2, wn = wid & 3;
    int m0 = blockIdx.x * 128 + wm * 64;
    int n0 = blockIdx.y * 256 + wn * 64;

    v8f zero = {0.f, 0.f, 0.f, 0.f, 0.f, 0.f, 0.f, 0.f};
    v8f acc[4][4];
#pragma unroll
    for (int i = 0; i < 4; i++)
#pragma unroll
        for (int j = 0; j < 4; j++) acc[i][j] = zero;

    for (int kb = 0; kb < KOUT; kb += 32) {
        const unsigned short* abase;
        int ald, ak;
        if (kb < HID) { abase = mlp_act; ald = HID; ak = kb; }
        else          { abase = attn;    ald = CC;  ak = kb - HID; }
        v16bf af[4], bf_[4];
#pragma unroll
        for (int i = 0; i < 4; i++) af[i]  = load_a(abase, ald, m0 + 16 * i, ak, lane);
#pragma unroll
        for (int j = 0; j < 4; j++) bf_[j] = load_b(wsum, KOUT, n0 + 16 * j, kb, lane);
#pragma unroll
        for (int i = 0; i < 4; i++)
#pragma unroll
            for (int j = 0; j < 4; j++) acc[i][j] = wmma_bf16(af[i], bf_[j], acc[i][j]);
    }

    int half = (lane >= 16) ? 8 : 0;
    int ncol = lane & 15;
#pragma unroll
    for (int j = 0; j < 4; j++) {
        int c = n0 + 16 * j + ncol;
        float bias = out_b[c] + out_b[c + CC];
        float g    = ls_gamma[c];
#pragma unroll
        for (int i = 0; i < 4; i++) {
#pragma unroll
            for (int r = 0; r < 8; r++) {
                int m = m0 + 16 * i + r + half;
                float v = acc[i][j][r] + bias;
                out[(size_t)m * CC + c] = x[(size_t)m * CC + c] + g * v;
            }
        }
    }
}

// ---------------------------------------------------------------------------
// Host-side launcher
// ---------------------------------------------------------------------------
extern "C" void kernel_launch(void* const* d_in, const int* in_sizes, int n_in,
                              void* d_out, int out_size, void* d_ws, size_t ws_size,
                              hipStream_t stream) {
    const float* x          = (const float*)d_in[0];
    const float* in_norm_w  = (const float*)d_in[1];
    const float* in_norm_b  = (const float*)d_in[2];
    const float* in_proj_w  = (const float*)d_in[3];
    const float* mlp_bias   = (const float*)d_in[4];
    const float* q_norm_w   = (const float*)d_in[5];
    const float* q_norm_b   = (const float*)d_in[6];
    const float* k_norm_w   = (const float*)d_in[7];
    const float* k_norm_b   = (const float*)d_in[8];
    const float* out_proj_w = (const float*)d_in[9];
    const float* out_proj_b = (const float*)d_in[10];
    const float* ls_gamma   = (const float*)d_in[11];
    float* out = (float*)d_out;

    char* wp = (char*)d_ws;
    size_t off = 0;
    auto take = [&](size_t bytes) -> char* {
        char* p = wp + off;
        off = (off + bytes + 255) & ~(size_t)255;
        return p;
    };
    unsigned short* Win  = (unsigned short*)take((size_t)OIN * CC * 2);   // 14.7 MB
    unsigned short* Wsum = (unsigned short*)take((size_t)CC * KOUT * 2);  // 10.5 MB
    unsigned short* Y    = (unsigned short*)take((size_t)MM * CC * 2);    // 16.8 MB
    unsigned short* MLP  = (unsigned short*)take((size_t)MM * HID * 2);   // 67.1 MB
    unsigned short* Qb   = (unsigned short*)take((size_t)MM * CC * 2);
    unsigned short* Kb   = (unsigned short*)take((size_t)MM * CC * 2);
    unsigned short* VT   = (unsigned short*)take((size_t)MM * CC * 2);
    unsigned short* QT   = (unsigned short*)take((size_t)MM * CC * 2);
    unsigned short* KT   = (unsigned short*)take((size_t)MM * CC * 2);
    unsigned short* ATT  = (unsigned short*)take((size_t)MM * CC * 2);

    cvt_bf16_kernel<<<(OIN * CC) / 256, 256, 0, stream>>>(in_proj_w, Win, OIN * CC);
    wsum_kernel<<<(CC * KOUT) / 256, 256, 0, stream>>>(out_proj_w, Wsum);
    ln_in_kernel<<<MM, 256, 0, stream>>>(x, in_norm_w, in_norm_b, Y);
    gemm_in_kernel<<<dim3(MM / 128, OIN / 256), 256, 0, stream>>>(Y, Win, mlp_bias,
                                                                  MLP, Qb, Kb, VT);
    qk_ln_kernel<<<(MM * HH * 2) / 8, 256, 0, stream>>>(Qb, Kb, q_norm_w, q_norm_b,
                                                        k_norm_w, k_norm_b, QT, KT);
    attn_kernel<<<BN * HH * (NN / 128), 256, 0, stream>>>(QT, KT, VT, ATT);
    gemm_out_kernel<<<dim3(MM / 128, CC / 256), 256, 0, stream>>>(MLP, ATT, Wsum,
                                                                  out_proj_b, ls_gamma,
                                                                  x, out);
}
